// LMSteinshark_66151086293381
// MI455X (gfx1250) — matrix operand order repaired and checked by reading, not compiled, not run on hardware
//
#include <hip/hip_runtime.h>
#include <hip/hip_bf16.h>
#include <math.h>

// ---------------------------------------------------------------------------
// Model constants (match the reference)
// ---------------------------------------------------------------------------
#define Lc 16
#define Bc 2
#define Nc 1024
#define Cc 512
#define Hc 16
#define Fc 1024
#define Vc 32768
#define DK 32           // Cc / Hc

// GEMM tiling: block 128x128, K-step 32, 8 waves (4 M x 2 N), wave tile 32x64
#define BM 128
#define BN 128
#define BKK 32

typedef __bf16 bf16_t;
typedef __attribute__((ext_vector_type(16))) __bf16 v16bf;
typedef __attribute__((ext_vector_type(8)))  float  v8f;

enum { GF_BIAS = 1, GF_GELU = 2, GF_ADD = 4, GF_BF16OUT = 8 };

__device__ __forceinline__ unsigned short f2bf(float f) {
    unsigned u = __float_as_uint(f);
    u += 0x7FFFu + ((u >> 16) & 1u);   // round-to-nearest-even
    return (unsigned short)(u >> 16);
}

// async DMA: global memory -> LDS, 16 bytes per lane, tracked by ASYNCcnt.
// lds = LDS byte offset (low 32 bits of flat shared address), g = global ptr.
__device__ __forceinline__ void async_ld_b128(unsigned lds, const void* g) {
    asm volatile("global_load_async_to_lds_b128 %0, %1, off"
                 :: "v"(lds), "v"(g) : "memory");
}
__device__ __forceinline__ void wait_async0() {
    asm volatile("s_wait_asynccnt 0x0" ::: "memory");
}

// ---------------------------------------------------------------------------
// WMMA bf16 GEMM:  C[M,N] = epilogue( alpha * A[M,K] @ Bt[N,K]^T + bias )
// A, Bt are bf16 (stored as u16). Bt is the weight in [N][K] ("transposed")
// layout so each lane's B fragment is one contiguous 32B read.
// Requirements: M % 128 == 0, K % 32 == 0 (true for every call here);
// N arbitrary (tile rows clamped on load, stores guarded).
// Staging is done with GLOBAL_LOAD_ASYNC_TO_LDS_B128 (no staging VGPRs),
// double-buffered in LDS, one s_wait_asynccnt + barrier per k-step.
// ---------------------------------------------------------------------------
__global__ __launch_bounds__(256, 1)
void gemm_bf16_wmma(const unsigned short* __restrict__ A, int lda,
                    const unsigned short* __restrict__ Bt, int ldb,
                    const float* __restrict__ bias,
                    float* __restrict__ Cf, unsigned short* __restrict__ Cb,
                    int ldc, int M, int N, int K, float alpha, int flags)
{
    __shared__ unsigned short As[2][BM * BKK];   // 8 KB per buffer
    __shared__ unsigned short Bs[2][BN * BKK];   // 8 KB per buffer

    const int tid  = threadIdx.x;
    const int lane = tid & 31;
    const int wave = tid >> 5;          // 8 waves: 4 (M) x 2 (N)
    const int wm   = wave >> 1;
    const int wn   = wave & 1;
    const int bm0  = blockIdx.y * BM;
    const int bn0  = blockIdx.x * BN;
    const int l15  = lane & 15;
    const int half = lane >> 4;

    // staging: each thread DMAs 2 x 16B for A and 2 x 16B for B per k-step
    const int r0 = tid >> 2;            // 0..63
    const int c0 = (tid & 3) * 8;       // 0,8,16,24

    // per-thread global pointers, rows clamped (OOB rows feed discarded outputs)
    const unsigned short* pA[2];
    const unsigned short* pB[2];
    #pragma unroll
    for (int rr = 0; rr < 2; ++rr) {
        int ra = bm0 + rr * 64 + r0; ra = ra < M ? ra : M - 1;
        int rb = bn0 + rr * 64 + r0; rb = rb < N ? rb : N - 1;
        pA[rr] = A  + (size_t)ra * lda + c0;
        pB[rr] = Bt + (size_t)rb * ldb + c0;
    }

    // LDS byte offsets for this thread's staging slots (buffer stride 8192 B,
    // rr stride 64 rows * 32 elems * 2 B = 4096 B)
    const unsigned ldsA0 = (unsigned)(size_t)&As[0][r0 * BKK + c0];
    const unsigned ldsB0 = (unsigned)(size_t)&Bs[0][r0 * BKK + c0];

    // ---- prologue: async-stage k-slice 0 into buffer 0
    #pragma unroll
    for (int rr = 0; rr < 2; ++rr) {
        async_ld_b128(ldsA0 + rr * 4096, pA[rr]);
        async_ld_b128(ldsB0 + rr * 4096, pB[rr]);
    }
    wait_async0();
    __syncthreads();

    // fragment read bases (constant per thread)
    const int aBase0 = (wm * 32 + l15) * BKK + half * 16;  // tm=0 row
    const int bBase0 = (wn * 64 + l15) * BKK + half * 16;  // tn=0 col

    v8f acc[2][4] = {};
    int buf = 0;

    for (int k0 = 0; k0 < K; k0 += BKK) {
        const bool has_next = (k0 + BKK) < K;

        // prefetch k+2 slice toward cache (speculative)
        if (k0 + 2 * BKK < K) {
            __builtin_prefetch(pA[0] + 2 * BKK, 0, 3);
            __builtin_prefetch(pB[0] + 2 * BKK, 0, 3);
        }

        // async DMA of next slice into the alternate buffer (overlaps WMMAs)
        if (has_next) {
            const unsigned nb = (unsigned)(buf ^ 1) * 8192u;
            #pragma unroll
            for (int rr = 0; rr < 2; ++rr) {
                async_ld_b128(ldsA0 + nb + rr * 4096, pA[rr] + BKK);
                async_ld_b128(ldsB0 + nb + rr * 4096, pB[rr] + BKK);
            }
            pA[0] += BKK; pA[1] += BKK; pB[0] += BKK; pB[1] += BKK;
        }

        // ---- A fragments (live across both B halves)
        v16bf af[2];
        #pragma unroll
        for (int tm = 0; tm < 2; ++tm) {
            const unsigned short* p = &As[buf][aBase0 + tm * 16 * BKK];
            __builtin_memcpy(&af[tm], p, 32);                  // K 0-15 / 16-31
        }

        // ---- B fragments in two halves to cap live registers
        #pragma unroll
        for (int hN = 0; hN < 2; ++hN) {
            v16bf bf[2];
            #pragma unroll
            for (int j = 0; j < 2; ++j) {
                const int tn = hN * 2 + j;
                __builtin_memcpy(&bf[j], &Bs[buf][bBase0 + tn * 16 * BKK], 32);
            }
            #pragma unroll
            for (int tm = 0; tm < 2; ++tm)
                #pragma unroll
                for (int j = 0; j < 2; ++j)
                    acc[tm][hN * 2 + j] = __builtin_amdgcn_wmma_f32_16x16x32_bf16(
                        false, af[tm], false, bf[j],
                        (short)0, acc[tm][hN * 2 + j], false, false);
        }

        // ---- wait for this wave's async DMAs, then barrier; flip buffers
        if (has_next) {
            wait_async0();
            __syncthreads();
            buf ^= 1;
        }
    }

    // ---- epilogue: C/D layout -> row = elem + 8*half, col = lane&15
    #pragma unroll
    for (int tm = 0; tm < 2; ++tm) {
        #pragma unroll
        for (int tn = 0; tn < 4; ++tn) {
            int gcol = bn0 + wn * 64 + tn * 16 + l15;
            #pragma unroll
            for (int i = 0; i < 8; ++i) {
                int grow = bm0 + wm * 32 + tm * 16 + half * 8 + i;
                if (grow < M && gcol < N) {
                    float v = acc[tm][tn][i] * alpha;
                    if (flags & GF_BIAS) v += bias[gcol];
                    if (flags & GF_GELU) v = 0.5f * v * (1.0f + erff(v * 0.70710678118f));
                    size_t o = (size_t)grow * ldc + gcol;
                    if (flags & GF_BF16OUT)      Cb[o] = f2bf(v);
                    else if (flags & GF_ADD)     Cf[o] += v;
                    else                         Cf[o] = v;
                }
            }
        }
    }
}

// ---------------------------------------------------------------------------
// Helper kernels
// ---------------------------------------------------------------------------
__global__ void conv_f32_bf16(const float* __restrict__ in,
                              unsigned short* __restrict__ out, long n)
{
    long i = (long)blockIdx.x * blockDim.x + threadIdx.x;
    if (i < n) out[i] = f2bf(in[i]);
}

// in: [K][N] fp32 row-major  ->  out: [N][K] bf16
__global__ void tconv_f32_bf16(const float* __restrict__ in,
                               unsigned short* __restrict__ out, int K, int N)
{
    long i = (long)blockIdx.x * blockDim.x + threadIdx.x;
    if (i >= (long)K * N) return;
    int k = (int)(i / N), n = (int)(i % N);
    out[(size_t)n * K + k] = f2bf(in[i]);
}

__global__ void embed_kernel(const int* __restrict__ ids,
                             const float* __restrict__ temb,
                             const float* __restrict__ pemb,
                             float* __restrict__ x)
{
    long i = (long)blockIdx.x * blockDim.x + threadIdx.x;
    if (i >= (long)Bc * Nc * Cc) return;
    int t = (int)(i / Cc), c = (int)(i % Cc);
    int n = t % Nc;
    x[i] = temb[(size_t)ids[t] * Cc + c] + pemb[(size_t)n * Cc + c];
}

__global__ __launch_bounds__(128)
void layernorm_bf16(const float* __restrict__ x, const float* __restrict__ g,
                    const float* __restrict__ b, unsigned short* __restrict__ out)
{
    const int row = blockIdx.x, tid = threadIdx.x;
    const float* xr = x + (size_t)row * Cc;
    __shared__ float red[128];
    float s = 0.f;
    for (int c = tid; c < Cc; c += 128) s += xr[c];
    red[tid] = s; __syncthreads();
    for (int t = 64; t > 0; t >>= 1) { if (tid < t) red[tid] += red[tid + t]; __syncthreads(); }
    float mean = red[0] * (1.0f / Cc); __syncthreads();
    float v = 0.f;
    for (int c = tid; c < Cc; c += 128) { float d = xr[c] - mean; v += d * d; }
    red[tid] = v; __syncthreads();
    for (int t = 64; t > 0; t >>= 1) { if (tid < t) red[tid] += red[tid + t]; __syncthreads(); }
    float rstd = rsqrtf(red[0] * (1.0f / Cc) + 1e-5f);
    for (int c = tid; c < Cc; c += 128)
        out[(size_t)row * Cc + c] = f2bf((xr[c] - mean) * rstd * g[c] + b[c]);
}

// causal softmax over one score row; S fp32 [Nc][Nc] -> P bf16 [Nc][Nc]
__global__ __launch_bounds__(256)
void softmax_causal(const float* __restrict__ S, unsigned short* __restrict__ P)
{
    const int n = blockIdx.x, tid = threadIdx.x;
    __shared__ float red[256];
    const float* row = S + (size_t)n * Nc;
    float mx = -INFINITY;
    for (int m = tid; m < Nc; m += 256) if (m <= n) mx = fmaxf(mx, row[m]);
    red[tid] = mx; __syncthreads();
    for (int s = 128; s > 0; s >>= 1) { if (tid < s) red[tid] = fmaxf(red[tid], red[tid + s]); __syncthreads(); }
    mx = red[0]; __syncthreads();
    float sum = 0.f;
    for (int m = tid; m < Nc; m += 256) if (m <= n) sum += __expf(row[m] - mx);
    red[tid] = sum; __syncthreads();
    for (int s = 128; s > 0; s >>= 1) { if (tid < s) red[tid] += red[tid + s]; __syncthreads(); }
    float inv = 1.0f / red[0];
    for (int m = tid; m < Nc; m += 256) {
        float p = (m <= n) ? __expf(row[m] - mx) * inv : 0.0f;
        P[(size_t)n * Nc + m] = f2bf(p);
    }
}

// vt[b][h][dk][key] = qkv[(b*Nc+key)][2*Cc + h*DK + dk]
__global__ void extract_vt(const unsigned short* __restrict__ qkv,
                           unsigned short* __restrict__ vt)
{
    long i = (long)blockIdx.x * blockDim.x + threadIdx.x;
    if (i >= (long)Bc * Hc * DK * Nc) return;
    int key = (int)(i % Nc); long r = i / Nc;
    int d = (int)(r % DK); r /= DK;
    int h = (int)(r % Hc);
    int b = (int)(r / Hc);
    vt[i] = qkv[((size_t)(b * Nc + key)) * (3 * Cc) + 2 * Cc + h * DK + d];
}

__global__ void copy_targets(const int* __restrict__ t, int* __restrict__ out, int n)
{
    int i = blockIdx.x * blockDim.x + threadIdx.x;
    if (i < n) out[i] = t[i];
}

// ---------------------------------------------------------------------------
// Host orchestration
// ---------------------------------------------------------------------------
static inline size_t align256(size_t x) { return (x + 255) & ~(size_t)255; }

extern "C" void kernel_launch(void* const* d_in, const int* in_sizes, int n_in,
                              void* d_out, int out_size, void* d_ws, size_t ws_size,
                              hipStream_t stream)
{
    (void)in_sizes; (void)n_in; (void)out_size; (void)ws_size;

    const int*   input_ids = (const int*)  d_in[0];
    const int*   target    = (const int*)  d_in[1];
    const float* tok_emb   = (const float*)d_in[2];
    const float* pos_emb   = (const float*)d_in[3];
    const float* Wqkv      = (const float*)d_in[4];
    const float* bqkv      = (const float*)d_in[5];
    const float* Wo        = (const float*)d_in[6];
    const float* bo        = (const float*)d_in[7];
    const float* ln1_g     = (const float*)d_in[8];
    const float* ln1_b     = (const float*)d_in[9];
    const float* W1        = (const float*)d_in[10];
    const float* b1        = (const float*)d_in[11];
    const float* W2        = (const float*)d_in[12];
    const float* b2        = (const float*)d_in[13];
    const float* ln2_g     = (const float*)d_in[14];
    const float* ln2_b     = (const float*)d_in[15];
    const float* lnf_g     = (const float*)d_in[16];
    const float* lnf_b     = (const float*)d_in[17];

    // ---- workspace layout (bytes)
    char* ws = (char*)d_ws;
    size_t off = 0;
    auto alloc = [&](size_t bytes) { char* p = ws + off; off = align256(off + bytes); return p; };

    unsigned short* temb_bf = (unsigned short*)alloc((size_t)Vc * Cc * 2);
    float*          x       = (float*)         alloc((size_t)Bc * Nc * Cc * 4);
    unsigned short* hbuf    = (unsigned short*)alloc((size_t)Bc * Nc * Cc * 2);
    unsigned short* qkv     = (unsigned short*)alloc((size_t)Bc * Nc * 3 * Cc * 2);
    unsigned short* obuf    = (unsigned short*)alloc((size_t)Bc * Nc * Cc * 2);
    unsigned short* mid     = (unsigned short*)alloc((size_t)Bc * Nc * Fc * 2);
    float*          Sbuf    = (float*)         alloc((size_t)Nc * Nc * 4);
    unsigned short* Pbuf    = (unsigned short*)alloc((size_t)Nc * Nc * 2);
    unsigned short* vt      = (unsigned short*)alloc((size_t)Bc * Hc * DK * Nc * 2);
    unsigned short* wqkvT   = (unsigned short*)alloc((size_t)3 * Cc * Cc * 2);
    unsigned short* woT     = (unsigned short*)alloc((size_t)Cc * Cc * 2);
    unsigned short* w1T     = (unsigned short*)alloc((size_t)Fc * Cc * 2);
    unsigned short* w2T     = (unsigned short*)alloc((size_t)Cc * Fc * 2);

    const int M = Bc * Nc;   // 2048 tokens
    const float inv_sqrt_dk = 0.17677669529663687f; // 1/sqrt(32)

    auto gemm = [&](const void* A, int lda, const void* Bt, int ldb,
                    const float* bias, void* Cf, void* Cb, int ldc,
                    int Mm, int Nn, int Kk, float alpha, int flags) {
        dim3 grid((Nn + BN - 1) / BN, (Mm + BM - 1) / BM);
        gemm_bf16_wmma<<<grid, dim3(256), 0, stream>>>(
            (const unsigned short*)A, lda, (const unsigned short*)Bt, ldb,
            bias, (float*)Cf, (unsigned short*)Cb, ldc, Mm, Nn, Kk, alpha, flags);
    };

    // tok_emb -> bf16 (weight-tied logits B operand, already [V][C] = [N][K])
    {
        long n = (long)Vc * Cc;
        conv_f32_bf16<<<(unsigned)((n + 255) / 256), 256, 0, stream>>>(tok_emb, temb_bf, n);
    }
    // x = tok_emb[ids] + pos_emb
    {
        long n = (long)M * Cc;
        embed_kernel<<<(unsigned)((n + 255) / 256), 256, 0, stream>>>(input_ids, tok_emb, pos_emb, x);
    }

    for (int l = 0; l < Lc; ++l) {
        // weights -> bf16 [N][K] layout
        { long n = (long)Cc * 3 * Cc;
          tconv_f32_bf16<<<(unsigned)((n + 255) / 256), 256, 0, stream>>>(Wqkv + (size_t)l * Cc * 3 * Cc, wqkvT, Cc, 3 * Cc); }
        { long n = (long)Cc * Cc;
          tconv_f32_bf16<<<(unsigned)((n + 255) / 256), 256, 0, stream>>>(Wo + (size_t)l * Cc * Cc, woT, Cc, Cc); }
        { long n = (long)Cc * Fc;
          tconv_f32_bf16<<<(unsigned)((n + 255) / 256), 256, 0, stream>>>(W1 + (size_t)l * Cc * Fc, w1T, Cc, Fc); }
        { long n = (long)Fc * Cc;
          tconv_f32_bf16<<<(unsigned)((n + 255) / 256), 256, 0, stream>>>(W2 + (size_t)l * Fc * Cc, w2T, Fc, Cc); }

        // h = LN1(x)
        layernorm_bf16<<<M, 128, 0, stream>>>(x, ln1_g + (size_t)l * Cc, ln1_b + (size_t)l * Cc, hbuf);

        // qkv = h @ WqkvT + bqkv  (bf16 out, [M][3C])
        gemm(hbuf, Cc, wqkvT, Cc, bqkv + (size_t)l * 3 * Cc,
             nullptr, qkv, 3 * Cc, M, 3 * Cc, Cc, 1.0f, GF_BIAS | GF_BF16OUT);

        // Vt[b][h][dk][key]
        { long n = (long)Bc * Hc * DK * Nc;
          extract_vt<<<(unsigned)((n + 255) / 256), 256, 0, stream>>>(qkv, vt); }

        for (int b = 0; b < Bc; ++b) {
            for (int h = 0; h < Hc; ++h) {
                const unsigned short* Q = qkv + (size_t)(b * Nc) * 3 * Cc + h * DK;
                const unsigned short* Kp = qkv + (size_t)(b * Nc) * 3 * Cc + Cc + h * DK;
                // S = (Q @ K^T) / sqrt(dk)   [Nc][Nc] fp32
                gemm(Q, 3 * Cc, Kp, 3 * Cc, nullptr, Sbuf, nullptr, Nc,
                     Nc, Nc, DK, inv_sqrt_dk, 0);
                // P = causal softmax(S) (bf16)
                softmax_causal<<<Nc, 256, 0, stream>>>(Sbuf, Pbuf);
                // O = P @ V  -> write into obuf at column h*DK (merged heads)
                gemm(Pbuf, Nc, vt + (size_t)(b * Hc + h) * DK * Nc, Nc, nullptr,
                     nullptr, obuf + (size_t)(b * Nc) * Cc + h * DK, Cc,
                     Nc, DK, Nc, 1.0f, GF_BF16OUT);
            }
        }

        // x += O @ WoT + bo
        gemm(obuf, Cc, woT, Cc, bo + (size_t)l * Cc,
             x, nullptr, Cc, M, Cc, Cc, 1.0f, GF_BIAS | GF_ADD);

        // h = LN2(x)
        layernorm_bf16<<<M, 128, 0, stream>>>(x, ln2_g + (size_t)l * Cc, ln2_b + (size_t)l * Cc, hbuf);

        // mid = gelu(h @ W1T + b1)   (bf16, [M][F])
        gemm(hbuf, Cc, w1T, Cc, b1 + (size_t)l * Fc,
             nullptr, mid, Fc, M, Fc, Cc, 1.0f, GF_BIAS | GF_GELU | GF_BF16OUT);

        // x += mid @ W2T + b2
        gemm(mid, Fc, w2T, Fc, b2 + (size_t)l * Cc,
             x, nullptr, Cc, M, Cc, Fc, 1.0f, GF_BIAS | GF_ADD);
    }

    // final LN + weight-tied logits -> d_out (fp32 [B][N][V])
    layernorm_bf16<<<M, 128, 0, stream>>>(x, lnf_g, lnf_b, hbuf);
    gemm(hbuf, Cc, temb_bf, Cc, nullptr,
         d_out, nullptr, Vc, M, Vc, Cc, 1.0f, 0);

    // append target_ids after logits
    copy_targets<<<(M + 255) / 256, 256, 0, stream>>>(
        target, (int*)((float*)d_out + (size_t)M * Vc), M);
}
